// WFSA_40475771797975
// MI455X (gfx1250) — matrix operand order repaired and checked by reading, not compiled, not run on hardware
//
#include <hip/hip_runtime.h>
#include <stdint.h>

// ---------------------------------------------------------------------------
// WFSA scan -> parallel matrix-product scan with CDNA5 bf16 WMMA
//   B=32, T=1024, V=512, K=128, C=2
// Stage 1 (per 32-step chunk): trans[(tc,b)][i,k] = sum_v transits[i,k,v]*x[b,t,v]
//     GEMM [16384 x 512] x [512 x 1024]; B operand staged to LDS via
//     global_load_async_to_lds_b128 (+ s_wait_asynccnt), A streamed from L2.
// Stage 2: per-batch tree reduction of 32 transition matrices (bf16 WMMA),
//     parity-based row-major/transposed storage so both operands stream
//     contiguously.
// Stage 3: state = chunkP @ state, sequential over 32 chunks.
// Stage 4: out = state @ W_dec^T + b_dec
// ---------------------------------------------------------------------------

typedef __bf16 bf16_t;
typedef bf16_t bf16x8 __attribute__((ext_vector_type(8)));
typedef bf16_t v16bf  __attribute__((ext_vector_type(16)));
typedef float  v8f    __attribute__((ext_vector_type(8)));

#define KD 128          // state dim
#define VD 512          // vocab dim
#define BD 32           // batch
#define TD 1024         // time
#define TC 32           // chunk length
#define MATSZ (KD*KD)   // 16384 elements per transition matrix

// ---- WMMA fragment loaders (wave32; layouts per CDNA5 ISA 7.12.2) ----------

// A operand 16x32 bf16, row-major memory, leading dim ld (elements).
__device__ __forceinline__ v16bf load_frag_a(const bf16_t* __restrict__ base, int ld) {
  const int lane = threadIdx.x & 31;
  const int m = lane & 15, g = lane >> 4;
  const bf16_t* p = base + (size_t)m * ld + 8 * g;
  bf16x8 lo = *(const bf16x8*)p;
  bf16x8 hi = *(const bf16x8*)(p + 16);
  return __builtin_shufflevector(lo, hi, 0,1,2,3,4,5,6,7,8,9,10,11,12,13,14,15);
}

// B operand 32x16 bf16 where memory holds each column's K-run contiguously
// (memory is [col][k], leading dim ld). Works for global or LDS pointers.
template <typename P>
__device__ __forceinline__ v16bf load_frag_b(P base, int ld) {
  const int lane = threadIdx.x & 31;
  const int n = lane & 15, g = lane >> 4;
  P p = base + (size_t)n * ld + 8 * g;
  bf16x8 lo = *(const bf16x8*)p;
  bf16x8 hi = *(const bf16x8*)(p + 16);
  return __builtin_shufflevector(lo, hi, 0,1,2,3,4,5,6,7,8,9,10,11,12,13,14,15);
}

// ---- conversion / init kernels --------------------------------------------

__global__ void cvt_f32_bf16(const float* __restrict__ src, bf16_t* __restrict__ dst, long n) {
  long i = (long)blockIdx.x * blockDim.x + threadIdx.x;
  long stride = (long)gridDim.x * blockDim.x;
  for (; i < n; i += stride) dst[i] = (bf16_t)src[i];
}

// Xc[(tc*BD + b)*VD + v] = xs[b, t0+tc, v]  (bf16)
__global__ void cvt_xchunk(const float* __restrict__ xs, bf16_t* __restrict__ Xc, int t0) {
  int idx = blockIdx.x * blockDim.x + threadIdx.x;    // 0 .. TC*BD*VD-1
  int v = idx & (VD - 1);
  int col = idx >> 9;
  int tc = col >> 5, b = col & 31;
  Xc[idx] = (bf16_t)xs[((size_t)b * TD + t0 + tc) * VD + v];
}

__global__ void init_state(const float* __restrict__ init_vec, float* __restrict__ state) {
  state[blockIdx.x * KD + threadIdx.x] = init_vec[threadIdx.x];
}

// ---- Stage 1: trans GEMM with async-to-LDS B staging -----------------------
// grid (256, 16), block 128 (4 waves). All 4 waves share one 64-column x-slab
// (64 KB in LDS, staged with global_load_async_to_lds_b128); wave w computes
// M tile (blockIdx.x*4+w)*16 x 64 columns.
// Columns: col = tc*BD + b. Odd tc -> row-major [i*128+k], even tc -> [k*128+i].
__global__ void __launch_bounds__(128)
stage1_trans(const bf16_t* __restrict__ A,   // [16384, 512] bf16 row-major
             const bf16_t* __restrict__ Xc,  // [1024, 512]  bf16 row-major
             bf16_t* __restrict__ trans) {   // 1024 matrices of 16384
  __shared__ bf16_t btile[64 * VD];               // 64 KB: 64 cols x 512 v

  const int lane = threadIdx.x & 31;
  const int wid = threadIdx.x >> 5;
  const int mTile = (blockIdx.x * 4 + wid) * 16;  // row tile in [0,16384)
  const int ngroup = blockIdx.y;                  // 0..15, 64 columns each

  // ---- async stage: 64 KB slab Xc[ngroup*64 .. +63][0..511] -> LDS --------
  {
    const uint64_t gbase = (uint64_t)(uintptr_t)(Xc + (size_t)ngroup * 64 * VD);
    const uint32_t lbase = (uint32_t)(uintptr_t)&btile[0];   // LDS byte offset
#pragma unroll 4
    for (int it = 0; it < 32; ++it) {             // 128 thr * 32 it * 16 B = 64 KB
      const uint32_t off = (uint32_t)(threadIdx.x + it * 128) * 16u;
      const uint32_t ldsa = lbase + off;
      const uint64_t ga = gbase + off;
      asm volatile("global_load_async_to_lds_b128 %0, %1, off"
                   :: "v"(ldsa), "v"(ga) : "memory");
    }
    asm volatile("s_wait_asynccnt 0x0" ::: "memory");
  }
  __syncthreads();

  // ---- GEMM: 16 k-steps, A from global (L2-resident), B from LDS ----------
  v8f acc[4] = {};
  for (int kb = 0; kb < VD; kb += 32) {
    const bf16_t* aptr = A + (size_t)mTile * VD + kb;
    __builtin_prefetch(aptr + 32, 0, 1);          // global_prefetch next A slab
    v16bf af = load_frag_a(aptr, VD);
#pragma unroll
    for (int s = 0; s < 4; ++s) {
      v16bf bf = load_frag_b<const bf16_t*>(&btile[(size_t)(s * 16) * VD + kb], VD);
      acc[s] = __builtin_amdgcn_wmma_f32_16x16x32_bf16(
          false, af, false, bf, (short)0, acc[s], false, false);
    }
  }

  // ---- parity store --------------------------------------------------------
  const int n = lane & 15, g = lane >> 4;
#pragma unroll
  for (int s = 0; s < 4; ++s) {
    const int nTile = ngroup * 64 + s * 16;
    const int col = nTile + n;                    // = tc*32 + b
    const int tc = nTile >> 5;                    // uniform across the tile
    bf16_t* mat = trans + (size_t)col * MATSZ;
    if (tc & 1) {
      // row-major [i*128+k] == offset m : contiguous 8 per lane
      bf16x8 v;
#pragma unroll
      for (int r = 0; r < 8; ++r) v[r] = (bf16_t)acc[s][r];
      *(bf16x8*)(mat + mTile + 8 * g) = v;
    } else {
      // transposed [k*128+i] : i fixed for whole tile
      const int i0 = mTile >> 7;
      const int k0 = (mTile & 127) + 8 * g;
#pragma unroll
      for (int r = 0; r < 8; ++r) mat[(size_t)(k0 + r) * KD + i0] = (bf16_t)acc[s][r];
    }
  }
}

// ---- Stage 2: pairwise matrix product C_j = M_{2j+1} @ M_{2j} --------------
// in: matrix q = j*32 + b; odd j stored row-major (A side), even j transposed (B side).
// grid (npairs*32, 8), block 256 (8 waves = 8 column tiles).
__global__ void reduce_pairs(const bf16_t* __restrict__ in, bf16_t* __restrict__ out) {
  const int lane = threadIdx.x & 31;
  const int w = threadIdx.x >> 5;                 // tile column 0..7
  const int j = blockIdx.x >> 5;
  const int b = blockIdx.x & 31;
  const int iTile = blockIdx.y * 16;
  const int nTile = w * 16;

  const bf16_t* Am = in + ((size_t)(2 * j + 1) * BD + b) * MATSZ;  // row-major [i][k]
  const bf16_t* Bm = in + ((size_t)(2 * j) * BD + b) * MATSZ;      // transposed [n][k]

  v8f acc = {};
#pragma unroll
  for (int kb = 0; kb < KD; kb += 32) {
    v16bf af = load_frag_a(Am + (size_t)iTile * KD + kb, KD);
    v16bf bf = load_frag_b<const bf16_t*>(Bm + (size_t)nTile * KD + kb, KD);
    acc = __builtin_amdgcn_wmma_f32_16x16x32_bf16(
        false, af, false, bf, (short)0, acc, false, false);
  }

  bf16_t* Cm = out + ((size_t)j * BD + b) * MATSZ;
  const int n = lane & 15, g = lane >> 4;
  if (j & 1) {
    // next level uses as A: row-major [i*128 + n]
#pragma unroll
    for (int r = 0; r < 8; ++r)
      Cm[(size_t)(iTile + 8 * g + r) * KD + nTile + n] = (bf16_t)acc[r];
  } else {
    // next level uses as B: transposed [n*128 + i] -> contiguous per lane
    bf16x8 v;
#pragma unroll
    for (int r = 0; r < 8; ++r) v[r] = (bf16_t)acc[r];
    *(bf16x8*)(Cm + (size_t)(nTile + n) * KD + iTile + 8 * g) = v;
  }
}

// ---- Stage 3: state = chunkP @ state ---------------------------------------
// P stored transposed: mem[k*128 + i] = P[i][k]. grid BD, block KD.
__global__ void chunk_apply(const bf16_t* __restrict__ P, float* __restrict__ state) {
  const int b = blockIdx.x, i = threadIdx.x;
  __shared__ float s[KD];
  s[i] = state[(size_t)b * KD + i];
  __syncthreads();
  const bf16_t* M = P + (size_t)b * MATSZ;
  float acc = 0.f;
#pragma unroll 8
  for (int k = 0; k < KD; ++k) acc += (float)M[(size_t)k * KD + i] * s[k];
  state[(size_t)b * KD + i] = acc;
}

// ---- Stage 4: decode -------------------------------------------------------
__global__ void decode(const float* __restrict__ W, const float* __restrict__ bias,
                       const float* __restrict__ state, float* __restrict__ out) {
  const int tid = threadIdx.x;       // 64 threads
  const int b = tid >> 1, c = tid & 1;
  float acc = bias[c];
  for (int k = 0; k < KD; ++k) acc += W[c * KD + k] * state[(size_t)b * KD + k];
  out[b * 2 + c] = acc;
}

// ---------------------------------------------------------------------------

extern "C" void kernel_launch(void* const* d_in, const int* in_sizes, int n_in,
                              void* d_out, int out_size, void* d_ws, size_t ws_size,
                              hipStream_t stream) {
  (void)in_sizes; (void)n_in; (void)out_size; (void)ws_size;
  const float* xs       = (const float*)d_in[0];  // [32,1024,512]
  const float* init_vec = (const float*)d_in[1];  // [128]
  const float* transits = (const float*)d_in[2];  // [128,128,512]
  const float* W_dec    = (const float*)d_in[3];  // [2,128]
  const float* b_dec    = (const float*)d_in[4];  // [2]
  float* out = (float*)d_out;                     // [32,2]

  char* ws = (char*)d_ws;                         // ~73 MB total, 256B-aligned slices
  bf16_t* A_bf  = (bf16_t*)(ws);                  // 16,777,216 B : transits bf16
  bf16_t* Xc    = (bf16_t*)(ws + 16777216);       //  1,048,576 B : chunk of xs, bf16
  bf16_t* trans = (bf16_t*)(ws + 17825792);       // 33,554,432 B : 1024 transition mats
  bf16_t* red0  = (bf16_t*)(ws + 51380224);       // 16,777,216 B
  bf16_t* red1  = (bf16_t*)(ws + 68157440);       //  8,388,608 B
  float*  state = (float*)(ws + 76546048);        //     16,384 B

  const long nA = (long)MATSZ * VD;               // 8,388,608
  cvt_f32_bf16<<<4096, 256, 0, stream>>>(transits, A_bf, nA);
  init_state<<<BD, KD, 0, stream>>>(init_vec, state);

  for (int c = 0; c < TD / TC; ++c) {             // 32 sequential chunks
    cvt_xchunk<<<(TC * BD * VD) / 256, 256, 0, stream>>>(xs, Xc, c * TC);
    stage1_trans<<<dim3(MATSZ / 64, 16), 128, 0, stream>>>(A_bf, Xc, trans);
    // tree reduce 32 -> 1 matrices per batch
    reduce_pairs<<<dim3(16 * BD, 8), 256, 0, stream>>>(trans, red0);
    reduce_pairs<<<dim3(8 * BD, 8), 256, 0, stream>>>(red0, red1);
    reduce_pairs<<<dim3(4 * BD, 8), 256, 0, stream>>>(red1, red0);
    reduce_pairs<<<dim3(2 * BD, 8), 256, 0, stream>>>(red0, red1);
    reduce_pairs<<<dim3(1 * BD, 8), 256, 0, stream>>>(red1, red0);
    chunk_apply<<<BD, KD, 0, stream>>>(red0, state);
  }

  decode<<<1, 64, 0, stream>>>(W_dec, b_dec, state, out);
}
